// PraxisByteLatentEncoder_51393578664481
// MI455X (gfx1250) — matrix operand order repaired and checked by reading, not compiled, not run on hardware
//
#include <hip/hip_runtime.h>
#include <hip/hip_bf16.h>
#include <math.h>

// ---------------- model constants ----------------
#define B_      2
#define S_      2048
#define D_      1024
#define DFF_    2816
#define M_      (B_ * S_)      // 4096 rows
#define WINDOW_ 512
#define NQT_    (S_ / 16)      // 128 query tiles per batch
#define NREL_   33             // key tiles per query tile band (512/16 + 1)
#define NRELP_  34             // padded (even) for K=32 WMMA steps
#define SPACE_  36
#define HASHV_  32000
#define OUTV_   32000

typedef _Float16 f16;
typedef __attribute__((ext_vector_type(16))) _Float16 v16h;
typedef __attribute__((ext_vector_type(8)))  _Float16 f16x8;
typedef __attribute__((ext_vector_type(8)))  float    v8f;

#define LDSPAD 40   // halves per LDS row (80B): 20-bank stride -> conflict-free b128

// =====================================================================
// Tiled GEMM: C[M,N] = A[M,K] @ W[K,N], fp32 in/out, f16 WMMA compute.
// Block = 128 threads = 4 waves. Block tile 128x64; wave tile 32x64
// (2x4 WMMA C tiles). K step 32. M%128==0, N%64==0, K%32==0.
// A tile row-major in LDS (frag K-chunks contiguous); B tile stored
// TRANSPOSED (BsT[n][k]) so B fragments are contiguous ds_load_b128.
// =====================================================================
__global__ __launch_bounds__(128) void gemm_f16_kernel(
    const float* __restrict__ A, const float* __restrict__ W,
    float* __restrict__ C, int N, int K)
{
  __shared__ f16 As[128][LDSPAD];
  __shared__ f16 BsT[64][LDSPAD];
  const int tid  = threadIdx.x;
  const int wave = tid >> 5;
  const int lane = tid & 31;
  const int m0 = blockIdx.y * 128;
  const int n0 = blockIdx.x * 64;

  v8f acc[2][4] = {};

  // A loader: thread t owns row (m0+t), loads 32 floats (8 float4)
  // B loader: thread t owns column n = t%64, k-half kh = t/64 (16 k's)
  const int bn = tid & 63;
  const int kh = (tid >> 6) * 16;

  // per-wave fragment addressing (wave32 WMMA layouts, ISA 7.12.2)
  const int mrow0  = (wave << 5) + (lane & 15);      // wave tile rows 32w..32w+31
  const int kbA    = (lane < 16) ? 0 : 8;            // A: elems 0..7 -> kbA.., 8..15 -> kbA+16..
  const int kbB    = (lane < 16) ? 0 : 8;            // B: elems 0..7 -> kbB.., 8..15 -> kbB+16..
  const int col    = lane & 15;
  const int rowoff = (lane < 16) ? 0 : 8;            // C: vgpr r -> M = r + rowoff

  for (int k0 = 0; k0 < K; k0 += 32) {
    if (k0 + 32 < K) {  // prefetch next K tile (global_prefetch_b8)
      __builtin_prefetch(&A[(size_t)(m0 + tid) * K + k0 + 32], 0, 1);
      __builtin_prefetch(&W[(size_t)(k0 + 32 + kh) * N + n0 + bn], 0, 1);
    }
    // ---- A tile: global float4 loads -> f16, contiguous b128 LDS stores ----
    {
      const float4* ap = (const float4*)&A[(size_t)(m0 + tid) * K + k0];
      #pragma unroll
      for (int c = 0; c < 4; ++c) {            // 4 chunks of 8 halves
        float4 x0 = ap[2 * c + 0];
        float4 x1 = ap[2 * c + 1];
        f16x8 h;
        h[0] = (f16)x0.x; h[1] = (f16)x0.y; h[2] = (f16)x0.z; h[3] = (f16)x0.w;
        h[4] = (f16)x1.x; h[5] = (f16)x1.y; h[6] = (f16)x1.z; h[7] = (f16)x1.w;
        *(f16x8*)&As[tid][8 * c] = h;
      }
    }
    // ---- B tile: coalesced per-k row loads, transpose in regs, b128 stores ----
    {
      f16x8 h0, h1;
      #pragma unroll
      for (int j = 0; j < 8; ++j)
        h0[j] = (f16)W[(size_t)(k0 + kh + j) * N + n0 + bn];
      #pragma unroll
      for (int j = 0; j < 8; ++j)
        h1[j] = (f16)W[(size_t)(k0 + kh + 8 + j) * N + n0 + bn];
      *(f16x8*)&BsT[bn][kh]     = h0;
      *(f16x8*)&BsT[bn][kh + 8] = h1;
    }
    __syncthreads();

    // ---- fragments: all contiguous ds_load_b128 ----
    v16h a0, a1;
    {
      f16x8 lo = *(const f16x8*)&As[mrow0][kbA];
      f16x8 hi = *(const f16x8*)&As[mrow0][kbA + 16];
      #pragma unroll
      for (int t = 0; t < 8; ++t) { a0[t] = lo[t]; a0[8 + t] = hi[t]; }
      lo = *(const f16x8*)&As[mrow0 + 16][kbA];
      hi = *(const f16x8*)&As[mrow0 + 16][kbA + 16];
      #pragma unroll
      for (int t = 0; t < 8; ++t) { a1[t] = lo[t]; a1[8 + t] = hi[t]; }
    }
    #pragma unroll
    for (int nt = 0; nt < 4; ++nt) {
      v16h b;
      f16x8 lo = *(const f16x8*)&BsT[16 * nt + col][kbB];
      f16x8 hi = *(const f16x8*)&BsT[16 * nt + col][kbB + 16];
      #pragma unroll
      for (int t = 0; t < 8; ++t) { b[t] = lo[t]; b[8 + t] = hi[t]; }
      acc[0][nt] = __builtin_amdgcn_wmma_f32_16x16x32_f16(false, a0, false, b, (short)0, acc[0][nt], false, false);
      acc[1][nt] = __builtin_amdgcn_wmma_f32_16x16x32_f16(false, a1, false, b, (short)0, acc[1][nt], false, false);
    }
    __syncthreads();
  }

  #pragma unroll
  for (int mt = 0; mt < 2; ++mt)
    #pragma unroll
    for (int r = 0; r < 8; ++r) {
      float* cp = &C[(size_t)(m0 + (wave << 5) + 16 * mt + rowoff + r) * N + n0 + col];
      #pragma unroll
      for (int nt = 0; nt < 4; ++nt)
        cp[16 * nt] = acc[mt][nt][r];
    }
}

// =====================================================================
// Hash n-gram embeddings: emb = tok_emb[t] + sum_f hash_emb[f][h_f % 32000]
// =====================================================================
__global__ __launch_bounds__(256) void hash_embed_kernel(
    const int* __restrict__ ids, const float* __restrict__ tok,
    const float* __restrict__ hemb, float* __restrict__ out)
{
  const int g = blockIdx.x;                 // b*S + s
  const int b = g / S_, s = g % S_;
  const int* tb = &ids[b * S_];
  const unsigned t0 = (unsigned)tb[s];
  const unsigned t1 = (s >= 1) ? (unsigned)tb[s - 1] : 0u;
  const unsigned t2 = (s >= 2) ? (unsigned)tb[s - 2] : 0u;
  const unsigned t3 = (s >= 3) ? (unsigned)tb[s - 3] : 0u;
  const unsigned mults[3] = {2654435761u, 805459861u, 2246822519u};
  const float* e0 = &tok[(size_t)tb[s] * D_];
  const float* ef[3];
  #pragma unroll
  for (int f = 0; f < 3; ++f) {
    unsigned a = mults[f];
    unsigned h = ((t0 * a + t1) * a + t2) * a + t3;
    ef[f] = &hemb[((size_t)f * HASHV_ + (h % (unsigned)HASHV_)) * D_];
  }
  for (int d = threadIdx.x; d < D_; d += 256)
    out[(size_t)g * D_ + d] = e0[d] + ef[0][d] + ef[1][d] + ef[2][d];
}

// patch boundaries: patch_ids is a cumsum => segments are contiguous runs
__global__ void patch_meta_kernel(const int* __restrict__ ids,
                                  int* __restrict__ patch_ids,
                                  int* __restrict__ seg_starts,
                                  int* __restrict__ nseg)
{
  const int b = blockIdx.x;
  const int* tb = &ids[b * S_];
  int* ss = &seg_starts[b * (S_ + 1)];
  int cnt = 0;
  for (int i = 0; i < S_; ++i) {
    if (i == 0 || tb[i - 1] == SPACE_) ss[cnt++] = i;
    patch_ids[b * S_ + i] = cnt - 1;
  }
  ss[cnt] = S_;
  nseg[b] = cnt;
}

// =====================================================================
// RMSNorm: one block (256 thr) per row of D_
// =====================================================================
__global__ __launch_bounds__(256) void rmsnorm_kernel(
    const float* __restrict__ x, const float* __restrict__ w, float* __restrict__ y)
{
  __shared__ float red[8];
  __shared__ float scale_s;
  const size_t row = blockIdx.x;
  const float* xr = &x[row * D_];
  float s = 0.f;
  for (int d = threadIdx.x; d < D_; d += 256) { float v = xr[d]; s += v * v; }
  for (int off = 16; off > 0; off >>= 1) s += __shfl_xor(s, off);
  if ((threadIdx.x & 31) == 0) red[threadIdx.x >> 5] = s;
  __syncthreads();
  if (threadIdx.x == 0) {
    float tot = 0.f;
    for (int i = 0; i < 8; ++i) tot += red[i];
    scale_s = rsqrtf(tot * (1.0f / D_) + 1e-5f);
  }
  __syncthreads();
  const float sc = scale_s;
  for (int d = threadIdx.x; d < D_; d += 256)
    y[row * D_ + d] = xr[d] * sc * w[d];
}

// RoPE in-place, one block per (b,s) row, dh = D_ (single head), half = 512
__global__ __launch_bounds__(256) void rope_kernel(float* __restrict__ x)
{
  const size_t row = blockIdx.x;
  const int s = (int)(row % S_);
  float* xr = &x[row * D_];
  const float nlog = -9.210340371976184f / 512.0f;   // -ln(10000)/half
  for (int i = threadIdx.x; i < 512; i += 256) {
    float freq = __expf((float)i * nlog);
    float sn, c;
    __sincosf((float)s * freq, &sn, &c);
    float x1 = xr[i], x2 = xr[i + 512];
    xr[i]       = x1 * c - x2 * sn;
    xr[i + 512] = x1 * sn + x2 * c;
  }
}

// =====================================================================
// Banded attention scores: one wave per (rel, qtile, b). 16x16 score tile
// over K=1024 via 32 WMMA steps, scaled + causal/window masked.
// Fragments loaded straight from global with explicit float4 (b128) loads.
// =====================================================================
__global__ __launch_bounds__(32) void attn_scores_kernel(
    const float* __restrict__ q, const float* __restrict__ k, float* __restrict__ sband)
{
  const int rel = blockIdx.x, qt = blockIdx.y, b = blockIdx.z;
  const int lane = threadIdx.x;
  const int n = lane & 15;
  const int rowoff = (lane < 16) ? 0 : 8;
  const int ktile = qt - (NREL_ - 1) + rel;
  float* out = &sband[(((size_t)b * NQT_ + qt) * NREL_ + rel) * 256];
  if (ktile < 0) {
    #pragma unroll
    for (int r = 0; r < 8; ++r) out[(rowoff + r) * 16 + n] = -1e30f;
    return;
  }
  const int kbA = (lane < 16) ? 0 : 16;
  const int kbB = (lane < 16) ? 0 : 16;
  const float* qp = &q[((size_t)b * S_ + qt * 16 + (lane & 15)) * D_];
  const float* kp = &k[((size_t)b * S_ + ktile * 16 + n) * D_];
  v8f acc = {};
  for (int k0 = 0; k0 < D_; k0 += 32) {
    v16h a, bb;
    {
      const float4* p0 = (const float4*)&qp[k0 + kbA];
      const float4* p1 = (const float4*)&qp[k0 + kbA + 8];
      float4 x0 = p0[0], x1 = p0[1], y0 = p1[0], y1 = p1[1];
      a[0] = (f16)x0.x; a[1] = (f16)x0.y; a[2]  = (f16)x0.z; a[3]  = (f16)x0.w;
      a[4] = (f16)x1.x; a[5] = (f16)x1.y; a[6]  = (f16)x1.z; a[7]  = (f16)x1.w;
      a[8] = (f16)y0.x; a[9] = (f16)y0.y; a[10] = (f16)y0.z; a[11] = (f16)y0.w;
      a[12] = (f16)y1.x; a[13] = (f16)y1.y; a[14] = (f16)y1.z; a[15] = (f16)y1.w;
    }
    {
      const float4* p = (const float4*)&kp[k0 + kbB];
      float4 x0 = p[0], x1 = p[1], x2 = p[2], x3 = p[3];
      bb[0]  = (f16)x0.x; bb[1]  = (f16)x0.y; bb[2]  = (f16)x0.z; bb[3]  = (f16)x0.w;
      bb[4]  = (f16)x1.x; bb[5]  = (f16)x1.y; bb[6]  = (f16)x1.z; bb[7]  = (f16)x1.w;
      bb[8]  = (f16)x2.x; bb[9]  = (f16)x2.y; bb[10] = (f16)x2.z; bb[11] = (f16)x2.w;
      bb[12] = (f16)x3.x; bb[13] = (f16)x3.y; bb[14] = (f16)x3.z; bb[15] = (f16)x3.w;
    }
    acc = __builtin_amdgcn_wmma_f32_16x16x32_f16(false, a, false, bb, (short)0, acc, false, false);
  }
  #pragma unroll
  for (int r = 0; r < 8; ++r) {
    const int i = qt * 16 + rowoff + r;
    const int j = ktile * 16 + n;
    float val = acc[r] * 0.03125f;                 // 1/sqrt(1024)
    if (!(j <= i && (i - j) < WINDOW_)) val = -1e30f;
    out[(rowoff + r) * 16 + n] = val;
  }
}

// softmax over the 528-entry band of one query row; emit f16 probs (+pad tile)
__global__ __launch_bounds__(32) void attn_softmax_kernel(
    const float* __restrict__ sband, f16* __restrict__ pband)
{
  const int g = blockIdx.x;                 // b*S + i
  const int b = g / S_, i = g % S_;
  const int qt = i >> 4, m = i & 15;
  const int lane = threadIdx.x;
  const float* srow = &sband[(((size_t)b * NQT_ + qt) * NREL_) * 256 + m * 16];
  float mx = -3e38f;
  for (int idx = lane; idx < 528; idx += 32)
    mx = fmaxf(mx, srow[(idx >> 4) * 256 + (idx & 15)]);
  for (int off = 16; off > 0; off >>= 1) mx = fmaxf(mx, __shfl_xor(mx, off));
  float sum = 0.f;
  for (int idx = lane; idx < 528; idx += 32)
    sum += __expf(srow[(idx >> 4) * 256 + (idx & 15)] - mx);
  for (int off = 16; off > 0; off >>= 1) sum += __shfl_xor(sum, off);
  const float inv = 1.0f / sum;
  f16* prow = &pband[(((size_t)b * NQT_ + qt) * NRELP_) * 256 + m * 16];
  for (int idx = lane; idx < 528; idx += 32)
    prow[(idx >> 4) * 256 + (idx & 15)] =
        (f16)(__expf(srow[(idx >> 4) * 256 + (idx & 15)] - mx) * inv);
  for (int idx = 528 + lane; idx < 544; idx += 32)   // pad tile for K=32 PV steps
    prow[(idx >> 4) * 256 + (idx & 15)] = (f16)0.f;
}

// banded P @ V: one wave per (nchunk, qtile, b); 17 WMMA steps over 544 keys
__global__ __launch_bounds__(32) void attn_pv_kernel(
    const f16* __restrict__ pband, const float* __restrict__ v, float* __restrict__ o)
{
  const int nc = blockIdx.x, qt = blockIdx.y, b = blockIdx.z;
  const int lane = threadIdx.x;
  const int m = lane & 15;
  const int col = lane & 15;
  const int kbA = (lane < 16) ? 0 : 16;
  const int kbB = (lane < 16) ? 0 : 16;
  const int rowoff = (lane < 16) ? 0 : 8;
  const f16* prow = &pband[(((size_t)b * NQT_ + qt) * NRELP_) * 256];
  v8f acc = {};
  for (int s = 0; s < 17; ++s) {
    v16h a, bb;
    {
      // kk = 32s + kbA + t (t<8) stays inside one 16x16 band tile; contiguous f16
      const int kk0 = 32 * s + kbA;
      const int kk1 = kk0 + 8;
      f16x8 lo = *(const f16x8*)&prow[(kk0 >> 4) * 256 + m * 16 + (kk0 & 15)];
      f16x8 hi = *(const f16x8*)&prow[(kk1 >> 4) * 256 + m * 16 + (kk1 & 15)];
      #pragma unroll
      for (int t = 0; t < 8; ++t) { a[t] = lo[t]; a[8 + t] = hi[t]; }
    }
    #pragma unroll
    for (int t = 0; t < 16; ++t) {
      const int kk = 32 * s + kbB + t;
      const int row = qt * 16 - WINDOW_ + kk;
      float x = (row >= 0 && row < S_) ? v[((size_t)b * S_ + row) * D_ + nc * 16 + col] : 0.f;
      bb[t] = (f16)x;
    }
    acc = __builtin_amdgcn_wmma_f32_16x16x32_f16(false, a, false, bb, (short)0, acc, false, false);
  }
  #pragma unroll
  for (int r = 0; r < 8; ++r)
    o[((size_t)b * S_ + qt * 16 + rowoff + r) * D_ + nc * 16 + col] = acc[r];
}

// ---------------- elementwise helpers ----------------
__global__ void add_kernel(const float* __restrict__ a, const float* __restrict__ b,
                           float* __restrict__ out, int n)
{
  int i = blockIdx.x * blockDim.x + threadIdx.x;
  if (i < n) out[i] = a[i] + b[i];
}

__global__ void silu_mul_kernel(float* __restrict__ g1, const float* __restrict__ g3, int n)
{
  int i = blockIdx.x * blockDim.x + threadIdx.x;
  if (i < n) {
    float x = g1[i];
    g1[i] = (x / (1.0f + __expf(-x))) * g3[i];
  }
}

// segment max over contiguous patch ranges
__global__ __launch_bounds__(128) void pool_kernel(
    const float* __restrict__ x, const int* __restrict__ seg_starts,
    const int* __restrict__ nseg, float* __restrict__ pooled)
{
  const int p = blockIdx.x, b = blockIdx.y;
  if (p >= nseg[b]) return;
  const int lo = seg_starts[b * (S_ + 1) + p];
  const int hi = seg_starts[b * (S_ + 1) + p + 1];
  for (int d = threadIdx.x; d < D_; d += 128) {
    float mx = -3e38f;
    for (int j = lo; j < hi; ++j)
      mx = fmaxf(mx, x[((size_t)b * S_ + j) * D_ + d]);
    pooled[((size_t)b * S_ + p) * D_ + d] = mx;
  }
}

// hd = embeds + pooled[patch_id]
__global__ __launch_bounds__(256) void gather_add_kernel(
    const float* __restrict__ embeds, const float* __restrict__ pooled,
    const int* __restrict__ patch_ids, float* __restrict__ h)
{
  const int g = blockIdx.x;
  const int b = g / S_;
  const int pid = patch_ids[g];
  for (int d = threadIdx.x; d < D_; d += 256)
    h[(size_t)g * D_ + d] = embeds[(size_t)g * D_ + d] +
                            pooled[((size_t)b * S_ + pid) * D_ + d];
}

// =====================================================================
// Host-side orchestration
// =====================================================================
extern "C" void kernel_launch(void* const* d_in, const int* in_sizes, int n_in,
                              void* d_out, int out_size, void* d_ws, size_t ws_size,
                              hipStream_t stream)
{
  (void)in_sizes; (void)n_in; (void)out_size; (void)ws_size;
  const int*   input_ids = (const int*)d_in[0];
  const float* tok_emb   = (const float*)d_in[1];
  const float* hash_emb  = (const float*)d_in[2];
  const float* W[2][7];   // [enc/dec][wq wk wv wo w1 w2 w3]
  const float* NW[2][3];  // [enc/dec][norm_attn norm_ffn norm_out]
  for (int L = 0; L < 2; ++L) {
    int base = 3 + L * 10;
    for (int j = 0; j < 7; ++j) W[L][j]  = (const float*)d_in[base + j];
    for (int j = 0; j < 3; ++j) NW[L][j] = (const float*)d_in[base + 7 + j];
  }
  const float* out_proj = (const float*)d_in[23];
  float* out = (float*)d_out;

  // ---- workspace carving ----
  char* wp = (char*)d_ws;
  auto alloc = [&](size_t bytes) -> void* {
    void* p = (void*)wp;
    wp += (bytes + 255) & ~(size_t)255;
    return p;
  };
  const size_t MD = (size_t)M_ * D_;
  float* embeds = (float*)alloc(MD * 4);
  float* h      = (float*)alloc(MD * 4);
  float* xn     = (float*)alloc(MD * 4);
  float* q      = (float*)alloc(MD * 4);       // reused as attention output
  float* kb     = (float*)alloc(MD * 4);
  float* vb     = (float*)alloc(MD * 4);       // reused as pooled
  float* tmp    = (float*)alloc(MD * 4);
  float* g1     = (float*)alloc((size_t)M_ * DFF_ * 4);
  float* g3     = (float*)alloc((size_t)M_ * DFF_ * 4);
  int* patch_ids  = (int*)alloc((size_t)M_ * 4);
  int* seg_starts = (int*)alloc((size_t)B_ * (S_ + 1) * 4);
  int* nseg       = (int*)alloc((size_t)B_ * 4);
  // attention bands alias the (temporally dead) g3 region:
  float* sband = g3;                                                    // 8.6 MB
  f16*   pband = (f16*)((char*)g3 + (size_t)B_ * NQT_ * NREL_ * 256 * 4);
  float* pooled = vb;

  auto gemm = [&](const float* A, const float* Wt, float* Cm, int N, int K) {
    dim3 grid(N / 64, M_ / 128);
    gemm_f16_kernel<<<grid, 128, 0, stream>>>(A, Wt, Cm, N, K);
  };
  auto ew_grid = [](size_t n) { return dim3((unsigned)((n + 255) / 256)); };

  // ---- embeddings + patch metadata ----
  hash_embed_kernel<<<M_, 256, 0, stream>>>(input_ids, tok_emb, hash_emb, embeds);
  patch_meta_kernel<<<B_, 1, 0, stream>>>(input_ids, patch_ids, seg_starts, nseg);

  // ---- transformer layer (shared encoder/decoder path) ----
  auto run_layer = [&](float* hin, const float* const* Wl, const float* const* Nl) {
    // attention sublayer
    rmsnorm_kernel<<<M_, 256, 0, stream>>>(hin, Nl[0], xn);
    gemm(xn, Wl[0], q,  D_, D_);
    gemm(xn, Wl[1], kb, D_, D_);
    gemm(xn, Wl[2], vb, D_, D_);
    rope_kernel<<<M_, 256, 0, stream>>>(q);
    rope_kernel<<<M_, 256, 0, stream>>>(kb);
    attn_scores_kernel<<<dim3(NREL_, NQT_, B_), 32, 0, stream>>>(q, kb, sband);
    attn_softmax_kernel<<<M_, 32, 0, stream>>>(sband, pband);
    attn_pv_kernel<<<dim3(D_ / 16, NQT_, B_), 32, 0, stream>>>(pband, vb, q); // q := attn out
    gemm(q, Wl[3], tmp, D_, D_);
    add_kernel<<<ew_grid(MD), 256, 0, stream>>>(hin, tmp, hin, (int)MD);
    // ffn sublayer
    rmsnorm_kernel<<<M_, 256, 0, stream>>>(hin, Nl[1], xn);
    gemm(xn, Wl[4], g1, DFF_, D_);   // w1
    gemm(xn, Wl[6], g3, DFF_, D_);   // w3
    silu_mul_kernel<<<ew_grid((size_t)M_ * DFF_), 256, 0, stream>>>(g1, g3, M_ * DFF_);
    gemm(g1, Wl[5], tmp, D_, DFF_);  // w2
    add_kernel<<<ew_grid(MD), 256, 0, stream>>>(hin, tmp, hin, (int)MD);
  };

  // ---- encoder ----
  hipMemcpyAsync(h, embeds, MD * 4, hipMemcpyDeviceToDevice, stream);
  run_layer(h, W[0], NW[0]);
  rmsnorm_kernel<<<M_, 256, 0, stream>>>(h, NW[0][2], xn);

  // ---- patch pooling + decoder input ----
  pool_kernel<<<dim3(S_, B_), 128, 0, stream>>>(xn, seg_starts, nseg, pooled);
  gather_add_kernel<<<M_, 256, 0, stream>>>(embeds, pooled, patch_ids, h);

  // ---- decoder ----
  run_layer(h, W[1], NW[1]);
  rmsnorm_kernel<<<M_, 256, 0, stream>>>(h, NW[1][2], xn);

  // ---- output projection (4096 x 32000 x 1024) ----
  gemm(xn, out_proj, out, OUTV_, D_);
}
